// EchelleModel_83064667504737
// MI455X (gfx1250) — matrix operand order repaired and checked by reading, not compiled, not run on hardware
//
#include <hip/hip_runtime.h>

// ---------------- problem constants ----------------
#define N_PIX   4194304
#define N_BINS  350000
#define TAPS    901
#define KPRE    240                 // pre-pad so (kt + K - 16m) never goes negative
#define KEXT    1408                // 240 + 1152 + spare, zero-padded kernel array
#define PAD_L   512
#define PAD_R   768
#define PADTOT  (PAD_L + N_PIX + PAD_R)   // 4,195,584  (divisible by 256)
#define BLK_OUT 8192                // outputs per block (8 waves * 1024)
#define WIN     9216                // LDS input window (f16 elems)
#define NTILE   36                  // K tiles of 32: covers 901 taps + 15*16 row shift

typedef __attribute__((ext_vector_type(16))) _Float16 v16h;
typedef __attribute__((ext_vector_type(8)))  _Float16 v8h;
typedef __attribute__((ext_vector_type(8)))  float    v8f;

// unaligned 32-byte f16 vector load (signal Hankel operand is 2B-aligned)
struct __attribute__((packed)) V16U { v16h v; };

// async LDS staging path (gfx1250 GLOBAL_LOAD_ASYNC_TO_LDS_*, ASYNCcnt),
// guarded so absence of the builtins falls back to sync staging.
#if defined(__has_builtin)
#  if __has_builtin(__builtin_amdgcn_global_load_async_to_lds_b128) && \
      __has_builtin(__builtin_amdgcn_s_wait_asynccnt)
#    define ASYNC_STAGE 1
#  endif
#endif
#ifndef ASYNC_STAGE
#  define ASYNC_STAGE 0
#endif

// exact parameter types per hipcc diagnostic:
//   'int __attribute__((__vector_size__(4*sizeof(int)))) __device__ *'
typedef int v4i __attribute__((vector_size(16)));
typedef __attribute__((address_space(1))) v4i GA1v4i;   // global src
typedef __attribute__((address_space(3))) v4i LA3v4i;   // LDS dst

// ---------------- kernel 1: build broadening kernels on device ----------------
__global__ __launch_bounds__(1024)
void prep_kernels(const float* __restrict__ lnsig, const float* __restrict__ lnvsini,
                  _Float16* __restrict__ kRot, _Float16* __restrict__ kGauss) {
    __shared__ float rot[TAPS];
    __shared__ float red[1024];
    const int tid = threadIdx.x;

    const float vsini = 0.9f + expf(lnvsini[0]);
    float partial = 0.0f;
    if (tid < TAPS) {
        const float g  = -4.5f + 0.01f * (float)tid;     // linspace(-4.5,4.5,901)
        const float x  = (299792.458f * g / 10500.0f) / vsini;
        const float x2 = fminf(x * x, 1.0f);
        const float r  = (x2 < 0.99999999f) ? 2.0f * sqrtf(1.0f - x2) : 0.0f;
        rot[tid] = r;
        partial  = r;
    }
    red[tid] = partial;
    __syncthreads();
    for (int s = 512; s > 0; s >>= 1) {
        if (tid < s) red[tid] += red[tid + s];
        __syncthreads();
    }
    const float inv   = 1.0f / red[0];
    const float sigma = 0.01f + expf(lnsig[0]);
    const float coef  = 0.01f / (sigma * sqrtf(2.0f * 3.1415926654f));
    const float ish2  = -0.5f / (sigma * sigma);

    for (int j = tid; j < KEXT; j += 1024) {
        const int i = j - KPRE;
        _Float16 rv = (_Float16)0.0f, gv = (_Float16)0.0f;
        if (i >= 0 && i < TAPS) {
            rv = (_Float16)(rot[i] * inv);
            const float g = -4.5f + 0.01f * (float)i;
            gv = (_Float16)(coef * expf(ish2 * g * g));
        }
        kRot[j]   = rv;
        kGauss[j] = gv;
    }
}

// ---------------- kernel 2: f32 -> zero-padded f16 signal; zero temp pads ----------------
__global__ __launch_bounds__(256)
void init_pad(const float* __restrict__ src, _Float16* __restrict__ inPad,
              _Float16* __restrict__ tmpPad) {
    const int idx = blockIdx.x * 256 + threadIdx.x;
    if (idx >= PADTOT) return;
    _Float16 v = (_Float16)0.0f;
    if (idx >= PAD_L && idx < PAD_L + N_PIX) v = (_Float16)src[idx - PAD_L];
    inPad[idx]  = v;
    tmpPad[idx] = (_Float16)0.0f;
}

// ---------------- kernel 3: zero bin accumulators ----------------
__global__ __launch_bounds__(256)
void zero_bins(float* __restrict__ sums, float* __restrict__ cnts) {
    const int i = blockIdx.x * 256 + threadIdx.x;
    if (i < N_BINS) { sums[i] = 0.0f; cnts[i] = 0.0f; }
}

// ---------------- kernel 4/5: 901-tap "same" conv via WMMA ----------------
// D[m,n] = out[o0 + 16m + n];  A[m,k] = Kext[kt + k - 16m] (kernel, shifted rows)
// B[k,n] = f[o0 - 450 + kt + k + n] (signal Hankel window, staged in LDS)
__global__ __launch_bounds__(256)
void conv_wmma(const _Float16* __restrict__ inPad,   // padded base (real data at +PAD_L)
               const _Float16* __restrict__ kext,    // KEXT zero-extended kernel
               _Float16* __restrict__ outp) {        // out[0] target
    __shared__ __align__(16) _Float16 sIn[WIN];
    __shared__ __align__(16) _Float16 sK[KEXT];
    const int tid = threadIdx.x;
    const int blk = blockIdx.x;

    // cooperative stage: window [blk*8192 - 456, +WIN) in padded coords (16B aligned)
    const size_t gstart = (size_t)blk * BLK_OUT + (size_t)(PAD_L - 456);
    const v8h* gin = (const v8h*)(inPad + gstart);
    const v8h* gkn = (const v8h*)kext;
    v8h* sInV = (v8h*)sIn;
    v8h* sKV  = (v8h*)sK;
#if ASYNC_STAGE
    for (int i = tid; i < WIN / 8; i += 256)
        __builtin_amdgcn_global_load_async_to_lds_b128(
            (GA1v4i*)(gin + i), (LA3v4i*)(sInV + i), 0, 0);
    for (int i = tid; i < KEXT / 8; i += 256)
        __builtin_amdgcn_global_load_async_to_lds_b128(
            (GA1v4i*)(gkn + i), (LA3v4i*)(sKV + i), 0, 0);
    __builtin_amdgcn_s_wait_asynccnt(0);
#else
    for (int i = tid; i < WIN / 8; i += 256)  sInV[i] = gin[i];
    for (int i = tid; i < KEXT / 8; i += 256) sKV[i]  = gkn[i];
#endif
    __syncthreads();

    const int wave = tid >> 5;
    const int lane = tid & 31;
    const int hi   = lane >> 4;     // lane half select
    const int mn   = lane & 15;     // A: row m   | B/D: col n

    v8f acc[4] = {};                // 4 output tiles = 1024 outputs per wave

    for (int t = 0; t < NTILE; ++t) {
        const int kt = t * 32;
        // A fragment: per-lane halves e -> K = 16hi + e (contiguous span)
        const int abase = kt + 16 * hi - 16 * mn + KPRE;  // multiple of 8 -> 16B aligned
        const v8h a0 = *(const v8h*)(sK + abase);
        const v8h a1 = *(const v8h*)(sK + abase + 8);
        const v16h A = __builtin_shufflevector(a0, a1,
                         0,1,2,3,4,5,6,7,8,9,10,11,12,13,14,15);
        #pragma unroll
        for (int g = 0; g < 4; ++g) {
            // B fragment: per-lane halves e -> K = 16hi + e (contiguous span)
            const int bbase = wave * 1024 + g * 256 + kt + 16 * hi + mn + 6;
            const v16h B = ((const V16U*)(sIn + bbase))->v;
            acc[g] = __builtin_amdgcn_wmma_f32_16x16x32_f16(
                         false, A, false, B, (short)0, acc[g], false, false);
        }
    }

    // D layout: VGPR v, lanes0-15 -> M=v, lanes16-31 -> M=v+8; N = lane&15
    const size_t ob0 = (size_t)blk * BLK_OUT + (size_t)wave * 1024;
    #pragma unroll
    for (int g = 0; g < 4; ++g) {
        const size_t ob = ob0 + (size_t)g * 256 + (size_t)mn;
        #pragma unroll
        for (int v = 0; v < 8; ++v)
            outp[ob + 16 * (v + 8 * hi)] = (_Float16)acc[g][v];
    }
}

// ---------------- kernel 6: sorted-segment run-length accumulate ----------------
__global__ __launch_bounds__(256)
void seg_accum(const _Float16* __restrict__ conv, const int* __restrict__ seg,
               float* __restrict__ sums, float* __restrict__ cnts) {
    const int c = blockIdx.x * 256 + threadIdx.x;      // N_PIX/32 threads
    const size_t base = (size_t)c * 32;
    int   cur = seg[base];
    float s = 0.0f, n = 0.0f;
    for (int e = 0; e < 32; ++e) {
        const int   id = seg[base + e];
        const float v  = (float)conv[base + e];
        if (id != cur) {
            atomicAdd(&sums[cur], s);
            atomicAdd(&cnts[cur], n);
            cur = id; s = 0.0f; n = 0.0f;
        }
        s += v; n += 1.0f;
    }
    atomicAdd(&sums[cur], s);
    atomicAdd(&cnts[cur], n);
}

// ---------------- kernel 7: means, clip, drop first/last bin ----------------
__global__ __launch_bounds__(256)
void finalize(const float* __restrict__ sums, const float* __restrict__ cnts,
              float* __restrict__ out) {
    const int i = blockIdx.x * 256 + threadIdx.x;
    if (i < N_BINS - 2) {
        const int b = i + 1;
        const float m = sums[b] / fmaxf(cnts[b], 1.0f);
        out[i] = fminf(fmaxf(m, 0.0f), 1.0f);
    }
}

// ---------------- launch ----------------
extern "C" void kernel_launch(void* const* d_in, const int* in_sizes, int n_in,
                              void* d_out, int out_size, void* d_ws, size_t ws_size,
                              hipStream_t stream) {
    (void)in_sizes; (void)n_in; (void)out_size; (void)ws_size;
    const float* hires = (const float*)d_in[0];
    const float* lnsig = (const float*)d_in[1];
    const float* lnvs  = (const float*)d_in[2];
    const int*   seg   = (const int*)d_in[3];

    char* ws = (char*)d_ws;
    size_t off = 0;
    auto alloc = [&](size_t bytes) {
        size_t o = off;
        off += (bytes + 255) & ~(size_t)255;
        return o;
    };
    _Float16* kRot   = (_Float16*)(ws + alloc((size_t)KEXT * 2));
    _Float16* kGauss = (_Float16*)(ws + alloc((size_t)KEXT * 2));
    _Float16* inPad  = (_Float16*)(ws + alloc((size_t)PADTOT * 2));
    _Float16* tmpPad = (_Float16*)(ws + alloc((size_t)PADTOT * 2));
    _Float16* convb  = (_Float16*)(ws + alloc((size_t)N_PIX * 2));
    float*    sums   = (float*)(ws + alloc((size_t)N_BINS * 4));
    float*    cnts   = (float*)(ws + alloc((size_t)N_BINS * 4));

    prep_kernels<<<1, 1024, 0, stream>>>(lnsig, lnvs, kRot, kGauss);
    init_pad<<<PADTOT / 256, 256, 0, stream>>>(hires, inPad, tmpPad);
    zero_bins<<<(N_BINS + 255) / 256, 256, 0, stream>>>(sums, cnts);
    // pass 1: rotational broadening  (writes interior of padded temp)
    conv_wmma<<<N_PIX / BLK_OUT, 256, 0, stream>>>(inPad, kRot, tmpPad + PAD_L);
    // pass 2: instrumental Gaussian
    conv_wmma<<<N_PIX / BLK_OUT, 256, 0, stream>>>(tmpPad, kGauss, convb);
    seg_accum<<<(N_PIX / 32) / 256, 256, 0, stream>>>(convb, seg, sums, cnts);
    finalize<<<(N_BINS - 2 + 255) / 256, 256, 0, stream>>>(sums, cnts, (float*)d_out);
}
